// EgdeConv_27195732918827
// MI455X (gfx1250) — matrix operand order repaired and checked by reading, not compiled, not run on hardware
//
#include <hip/hip_runtime.h>

#define N_AP   1024
#define N_UE   4096
#define NEDGE  524288
#define DCH    64

typedef __attribute__((ext_vector_type(16))) __bf16        v16bf;
typedef __attribute__((ext_vector_type(8)))  float         v8f;
typedef __attribute__((ext_vector_type(4)))  float         f32x4;
typedef __attribute__((ext_vector_type(4)))  unsigned int  u32x4;

union FragBf {
    v16bf          v;
    u32x4          q[2];
    unsigned short h[16];
};

// ---- helpers -------------------------------------------------------------

// native f32 -> bf16 (clang lowers fptrunc to the hw convert on gfx1250)
__device__ __forceinline__ unsigned short f2bf(float f) {
    return __builtin_bit_cast(unsigned short, (__bf16)f);
}

// order-preserving f32 <-> u32 so atomicMax(u32) == float max
__device__ __forceinline__ unsigned f2ord(float f) {
    unsigned u = __builtin_bit_cast(unsigned, f);
    return (u & 0x80000000u) ? ~u : (u | 0x80000000u);
}
__device__ __forceinline__ float ord2f(unsigned e) {
    unsigned u = (e & 0x80000000u) ? (e & 0x7FFFFFFFu) : ~e;
    return __builtin_bit_cast(float, u);
}

__device__ __forceinline__ v8f wmma_bf16(const FragBf& a, const FragBf& b, v8f c) {
    // D = A(16x32 bf16) * B(32x16 bf16) + C(16x16 f32)
    return __builtin_amdgcn_wmma_f32_16x16x32_bf16(
        /*neg_a=*/false, a.v, /*neg_b=*/false, b.v,
        /*c_mod=*/(short)0, c, /*reuse_a=*/false, /*reuse_b=*/false);
}

// A fragment (16x32 bf16) from a global f32 row.
// ISA layout: lanes 0-15 hold M=lane, K runs {kk..kk+7, kk+16..kk+23};
// lanes 16-31 hold the +8-shifted runs.
__device__ __forceinline__ void build_a_global(FragBf& f, const float* row, int kk, int hi8) {
    const float* p = row + kk + hi8;
    f32x4 a0 = *(const f32x4*)(p + 0);
    f32x4 a1 = *(const f32x4*)(p + 4);
    f32x4 b0 = *(const f32x4*)(p + 16);
    f32x4 b1 = *(const f32x4*)(p + 20);
    f.v[0]  = (__bf16)a0.x; f.v[1]  = (__bf16)a0.y; f.v[2]  = (__bf16)a0.z; f.v[3]  = (__bf16)a0.w;
    f.v[4]  = (__bf16)a1.x; f.v[5]  = (__bf16)a1.y; f.v[6]  = (__bf16)a1.z; f.v[7]  = (__bf16)a1.w;
    f.v[8]  = (__bf16)b0.x; f.v[9]  = (__bf16)b0.y; f.v[10] = (__bf16)b0.z; f.v[11] = (__bf16)b0.w;
    f.v[12] = (__bf16)b1.x; f.v[13] = (__bf16)b1.y; f.v[14] = (__bf16)b1.z; f.v[15] = (__bf16)b1.w;
}

// A fragment from an LDS bf16 row (64 cols, row base already includes m*64)
__device__ __forceinline__ void build_a_lds(FragBf& f, const unsigned short* rowbase, int kk, int hi8) {
    const unsigned short* p = rowbase + kk + hi8;
    f.q[0] = *(const u32x4*)(p);        // K = kk+hi8 .. +7
    f.q[1] = *(const u32x4*)(p + 16);   // K = kk+hi8+16 .. +23
}

// B fragment (32x16 bf16) from transposed LDS weights Wt[n][k] (k stride 128).
// Layout: N = lane&15; lanes 0-15 hold K={kk..kk+7, kk+16..kk+23}, lanes 16-31 the +8-shifted runs
// -> two 8-halfword runs = two ds_load_b128.
__device__ __forceinline__ void build_b_lds(FragBf& f, const unsigned short* Wt, int n, int kk, int hi16) {
    const unsigned short* p = Wt + n * 128 + kk + hi16 / 2;
    f.q[0] = *(const u32x4*)(p);
    f.q[1] = *(const u32x4*)(p + 16);
}

// ---- workspace init ------------------------------------------------------

__global__ void init_ws(unsigned* __restrict__ m1r1, unsigned* __restrict__ m2r1,
                        unsigned* __restrict__ cntr1,
                        unsigned* __restrict__ m1hb, unsigned* __restrict__ m2hb,
                        unsigned* __restrict__ cnthb) {
    const int i = blockIdx.x * 256 + threadIdx.x;
    const unsigned ONEG = f2ord(-1e30f);
    if (i < N_AP * DCH) { m1r1[i] = ONEG; m2r1[i] = ONEG; cntr1[i] = 0u; }
    if (i < N_UE * DCH) { m1hb[i] = ONEG; m2hb[i] = ONEG; cnthb[i] = 0u; }
}

// ---- fused pass kernel ---------------------------------------------------
// PASS 1: compute r1, hb; atomic seg-max into m1 buffers
// PASS 2: recompute; tie-count + second-max into cnt/m2 buffers
// PASS 3: recompute; build c1/c2/agg; final [agg|eh]@W3 and store out

template <int PASS>
__global__ __launch_bounds__(256)
void edge_pass(const float* __restrict__ x_ap, const float* __restrict__ edge_hid,
               const float* __restrict__ W1, const float* __restrict__ b1,
               const float* __restrict__ W2, const float* __restrict__ b2,
               const float* __restrict__ W3, const float* __restrict__ b3,
               const int* __restrict__ src, const int* __restrict__ dst,
               float* __restrict__ out,
               unsigned* __restrict__ m1r1, unsigned* __restrict__ m2r1,
               unsigned* __restrict__ cntr1,
               unsigned* __restrict__ m1hb, unsigned* __restrict__ m2hb,
               unsigned* __restrict__ cnthb) {
    __shared__ __align__(16) unsigned short sW1t[64 * 128];   // W1^T  [n][k]
    __shared__ __align__(16) unsigned short sW2t[64 * 128];   // W2^T  [n][k]
    __shared__ __align__(16) unsigned short sW3t[64 * 128];   // W3^T  [n][k]
    __shared__ __align__(16) unsigned short sAgg[8 * 16 * 64]; // per-wave agg tile

    const int tid  = threadIdx.x;
    const int lane = tid & 31;
    const int wave = tid >> 5;
    const int l15  = lane & 15;
    const int hi8  = (lane >= 16) ? 8 : 0;
    const int hi16 = hi8 * 2;

    // ---- stage weights (transposed, bf16) --------------------------------
    for (int idx = tid; idx < 64 * 128; idx += 256) {
        const int k = idx >> 6, n = idx & 63;          // W is [128][64] row-major
        sW1t[n * 128 + k] = f2bf(W1[idx]);
        sW2t[n * 128 + k] = f2bf(W2[idx]);
        if (PASS == 3) sW3t[n * 128 + k] = f2bf(W3[idx]);
    }
    __syncthreads();

    // ---- this wave's 16-edge tile ----------------------------------------
    const int ebase = blockIdx.x * 128 + wave * 16;
    const int erow  = ebase + l15;
    const int se    = src[erow];
    const int de    = dst[erow];

    // A fragments covering concat-K 0..127 (kk 0,32 = gathered x_ap; 64,96 = edge_hid)
    FragBf fa[4];
    build_a_global(fa[0], x_ap + (size_t)se * DCH, 0, hi8);
    build_a_global(fa[1], x_ap + (size_t)se * DCH, 32, hi8);
    build_a_global(fa[2], edge_hid + (size_t)erow * DCH, 0, hi8);
    build_a_global(fa[3], edge_hid + (size_t)erow * DCH, 32, hi8);

    const v8f vzero = {0.f, 0.f, 0.f, 0.f, 0.f, 0.f, 0.f, 0.f};
    v8f r1[4] = {vzero, vzero, vzero, vzero};
    v8f hb[4] = {vzero, vzero, vzero, vzero};
    v8f c2p[4] = {vzero, vzero, vzero, vzero};

    // r1 = [s|eh] @ W1
#pragma unroll
    for (int kk4 = 0; kk4 < 4; ++kk4) {
#pragma unroll
        for (int nt = 0; nt < 4; ++nt) {
            FragBf fb;
            build_b_lds(fb, sW1t, nt * 16 + l15, kk4 * 32, hi16);
            r1[nt] = wmma_bf16(fa[kk4], fb, r1[nt]);
        }
    }
    // hb = eh @ W2b   (W2 rows 64..127)
#pragma unroll
    for (int kk4 = 0; kk4 < 2; ++kk4) {
#pragma unroll
        for (int nt = 0; nt < 4; ++nt) {
            FragBf fb;
            build_b_lds(fb, sW2t, nt * 16 + l15, 64 + kk4 * 32, hi16);
            hb[nt] = wmma_bf16(fa[2 + kk4], fb, hb[nt]);
        }
    }
    if (PASS == 3) {
        // c2pre = s @ W2a   (W2 rows 0..63)
#pragma unroll
        for (int kk4 = 0; kk4 < 2; ++kk4) {
#pragma unroll
            for (int nt = 0; nt < 4; ++nt) {
                FragBf fb;
                build_b_lds(fb, sW2t, nt * 16 + l15, kk4 * 32, hi16);
                c2p[nt] = wmma_bf16(fa[kk4], fb, c2p[nt]);
            }
        }
    }

    // per-row segment ids for the 8 C-rows this lane owns
    int srow[8], drow[8];
#pragma unroll
    for (int r = 0; r < 8; ++r) {
        srow[r] = __shfl(se, r + hi8, 32);
        drow[r] = __shfl(de, r + hi8, 32);
    }

    // ---- epilogue over C elements (n = nt*16 + l15, m = r + hi8) ---------
#pragma unroll
    for (int nt = 0; nt < 4; ++nt) {
        const int n = nt * 16 + l15;
        const float bn1 = b1[n];
        const float bn2 = (PASS == 3) ? b2[n] : 0.f;
#pragma unroll
        for (int r = 0; r < 8; ++r) {
            const float r1v = fmaxf(r1[nt][r] + bn1, 0.f);   // relu(r1)
            const float hbv = hb[nt][r];                      // raw hb
            const int sb = srow[r] * DCH + n;
            const int db = drow[r] * DCH + n;
            if (PASS == 1) {
                atomicMax(&m1r1[sb], f2ord(r1v));
                atomicMax(&m1hb[db], f2ord(hbv));
            } else if (PASS == 2) {
                const float m1a = ord2f(m1r1[sb]);
                const bool ia = (r1v == m1a);
                if (ia) atomicAdd(&cntr1[sb], 1u);
                atomicMax(&m2r1[sb], f2ord(ia ? -1e30f : r1v));
                const float m1b = ord2f(m1hb[db]);
                const bool ib = (hbv == m1b);
                if (ib) atomicAdd(&cnthb[db], 1u);
                atomicMax(&m2hb[db], f2ord(ib ? -1e30f : hbv));
            } else {
                const float m1a = ord2f(m1r1[sb]);
                const bool ia = (r1v == m1a);
                const float c1 = (ia && cntr1[sb] <= 1u) ? ord2f(m2r1[sb]) : m1a;
                const float m1b = ord2f(m1hb[db]);
                const bool ib = (hbv == m1b);
                const float ex = (ib && cnthb[db] <= 1u) ? ord2f(m2hb[db]) : m1b;
                const float c2 = fmaxf(c2p[nt][r] + bn2 + ex, 0.f);
                const float agg = fmaxf(c1, c2);
                sAgg[wave * 1024 + (r + hi8) * 64 + n] = f2bf(agg);  // C->A relayout
            }
        }
    }

    if (PASS == 3) {
        __syncthreads();   // make per-wave sAgg stores visible cross-lane
        FragBf ga[2];
        const unsigned short* myrow = &sAgg[wave * 1024 + l15 * 64];
        build_a_lds(ga[0], myrow, 0, hi8);
        build_a_lds(ga[1], myrow, 32, hi8);

        v8f o[4] = {vzero, vzero, vzero, vzero};
#pragma unroll
        for (int kk4 = 0; kk4 < 4; ++kk4) {
#pragma unroll
            for (int nt = 0; nt < 4; ++nt) {
                FragBf fb;
                build_b_lds(fb, sW3t, nt * 16 + l15, kk4 * 32, hi16);
                const FragBf& A = (kk4 < 2) ? ga[kk4] : fa[kk4];
                o[nt] = wmma_bf16(A, fb, o[nt]);
            }
        }
#pragma unroll
        for (int nt = 0; nt < 4; ++nt) {
            const int n = nt * 16 + l15;
            const float bn3 = b3[n];
#pragma unroll
            for (int r = 0; r < 8; ++r) {
                out[(size_t)(ebase + r + hi8) * DCH + n] = fmaxf(o[nt][r] + bn3, 0.f);
            }
        }
    }
}

// ---- launcher ------------------------------------------------------------

extern "C" void kernel_launch(void* const* d_in, const int* in_sizes, int n_in,
                              void* d_out, int out_size, void* d_ws, size_t ws_size,
                              hipStream_t stream) {
    const float* x_ap     = (const float*)d_in[0];
    const float* edge_hid = (const float*)d_in[1];
    const float* W1 = (const float*)d_in[2];
    const float* b1 = (const float*)d_in[3];
    const float* W2 = (const float*)d_in[4];
    const float* b2 = (const float*)d_in[5];
    const float* W3 = (const float*)d_in[6];
    const float* b3 = (const float*)d_in[7];
    const int*  src = (const int*)d_in[8];
    const int*  dst = (const int*)d_in[9];
    float* out = (float*)d_out;

    // segment-state workspace (order-preserving u32 encodings), ~3.84 MB (L2 resident)
    unsigned* ws    = (unsigned*)d_ws;
    unsigned* m1r1  = ws;
    unsigned* m2r1  = m1r1 + N_AP * DCH;
    unsigned* cntr1 = m2r1 + N_AP * DCH;
    unsigned* m1hb  = cntr1 + N_AP * DCH;
    unsigned* m2hb  = m1hb + N_UE * DCH;
    unsigned* cnthb = m2hb + N_UE * DCH;

    const int segBlocks  = (N_UE * DCH + 255) / 256;     // covers both AP and UE ranges
    const int passBlocks = NEDGE / 128;                  // 16 edges per wave, 8 waves/block

    init_ws<<<segBlocks, 256, 0, stream>>>(m1r1, m2r1, cntr1, m1hb, m2hb, cnthb);
    edge_pass<1><<<passBlocks, 256, 0, stream>>>(x_ap, edge_hid, W1, b1, W2, b2, W3, b3,
                                                 src, dst, out, m1r1, m2r1, cntr1,
                                                 m1hb, m2hb, cnthb);
    edge_pass<2><<<passBlocks, 256, 0, stream>>>(x_ap, edge_hid, W1, b1, W2, b2, W3, b3,
                                                 src, dst, out, m1r1, m2r1, cntr1,
                                                 m1hb, m2hb, cnthb);
    edge_pass<3><<<passBlocks, 256, 0, stream>>>(x_ap, edge_hid, W1, b1, W2, b2, W3, b3,
                                                 src, dst, out, m1r1, m2r1, cntr1,
                                                 m1hb, m2hb, cnthb);
}
